// RecurrentLIFNode_56831007261291
// MI455X (gfx1250) — compile-verified, matrix operand order checked by reading, not run on hardware
//
#include <hip/hip_runtime.h>
#include <hip/hip_bf16.h>
#include <stdint.h>

// Problem constants (from reference): T,B,H,L
#define T_DIM 512
#define B_DIM 64
#define H_DIM 512
#define L_DIM 4
#define NBLOCKS 64          // 64 blocks x 2 waves = 128 waves = 128 (16x16) tiles
#define RSTRIDE 520         // padded LDS row stride (bf16): 260 dwords -> bank rotate 4/row

typedef __bf16 v16bf __attribute__((ext_vector_type(16)));
typedef __bf16 v8bf  __attribute__((ext_vector_type(8)));
typedef float  v8f   __attribute__((ext_vector_type(8)));
typedef float  v4f   __attribute__((ext_vector_type(4)));

// ---------------------------------------------------------------------------
// Fragment loaders, matching CDNA5 WMMA 16-bit A/B VGPR layouts:
//   lane<16 : K chunks {k0..k0+7} and {k0+16..k0+23} with k0 = 32*kb
//   lane>=16: same with k0 = 32*kb + 8
// A: lane row = M (0..15);  B: lane row = N (0..15).
// Works for both global (vmem) and LDS (ds) source pointers after inlining.
// ---------------------------------------------------------------------------
__device__ __forceinline__ v16bf load_frag_bf16(const __bf16* p) {
  v8bf c0 = *(const v8bf*)(p);        // K = k0 .. k0+7
  v8bf c1 = *(const v8bf*)(p + 16);   // K = k0+16 .. k0+23
  return __builtin_shufflevector(c0, c1, 0,1,2,3,4,5,6,7,8,9,10,11,12,13,14,15);
}

__device__ __forceinline__ v16bf load_frag_f32(const float* __restrict__ p) {
  v4f a0 = *(const v4f*)(p);
  v4f a1 = *(const v4f*)(p + 4);
  v4f a2 = *(const v4f*)(p + 16);
  v4f a3 = *(const v4f*)(p + 20);
  v16bf r;
#pragma unroll
  for (int i = 0; i < 4; ++i) {
    r[i]      = (__bf16)a0[i];
    r[i + 4]  = (__bf16)a1[i];
    r[i + 8]  = (__bf16)a2[i];
    r[i + 12] = (__bf16)a3[i];
  }
  return r;
}

// ---------------------------------------------------------------------------
// Device-wide barrier: monotonic arrival counter (no reset races).
// All NBLOCKS blocks are co-resident (64 tiny blocks), so spinning is safe.
// ---------------------------------------------------------------------------
__device__ __forceinline__ void grid_barrier(unsigned* bar, unsigned round) {
  __syncthreads();
  if (threadIdx.x == 0) {
    __threadfence();                       // publish this block's stores
    atomicAdd(bar, 1u);
    const unsigned target = round * NBLOCKS;
    while (*(volatile unsigned*)bar < target) {
      __builtin_amdgcn_s_sleep(1);
    }
    __threadfence();                       // acquire other blocks' stores
  }
  __syncthreads();
}

// ---------------------------------------------------------------------------
// Init: convert weights fp32 -> bf16, zero persistent state (lives in d_out
// tail, becomes final_states), zero barrier counter.
// ---------------------------------------------------------------------------
__global__ void lif_init_kernel(const float* __restrict__ Wf,
                                const float* __restrict__ Wi,
                                __bf16* __restrict__ wf16,
                                __bf16* __restrict__ wi16,
                                float* __restrict__ states,
                                unsigned* __restrict__ bar) {
  const int idx    = blockIdx.x * blockDim.x + threadIdx.x;
  const int stride = gridDim.x * blockDim.x;
  const int wElems = L_DIM * H_DIM * H_DIM;
  const int sElems = L_DIM * B_DIM * H_DIM;
  for (int i = idx; i < wElems; i += stride) {
    wf16[i] = (__bf16)Wf[i];
    wi16[i] = (__bf16)Wi[i];
  }
  for (int i = idx; i < sElems; i += stride) states[i] = 0.0f;
  if (idx < 64) bar[idx] = 0u;
}

// ---------------------------------------------------------------------------
// Persistent recurrence kernel. 64 blocks x 64 threads (2 waves each).
// Block b owns M tile m=b>>4 and N-tile pair npair=b&15; wave w handles
// n = npair*2 + w. Each wave computes BOTH gates of its 16x16 tile so the
// LIF cell update fuses per-lane.
//
// The block's full weight slice (32 N-columns x 512 K x 2 gates x 4 layers,
// bf16, padded rows) is staged ONCE into 260KB of the WGP's 320KB LDS and
// then re-read via ds_load_b128 for all 512 timesteps: the recurrence is
// latency-bound and LDS is the only level that holds the 512x-reused weights
// at register-adjacent latency. Activations (y) still flow through L2.
// ---------------------------------------------------------------------------
__global__ void __launch_bounds__(64)
lif_persistent_kernel(const float*  __restrict__ x,      // [T,B,H] f32
                      const __bf16* __restrict__ wf,     // [L,H,H] bf16
                      const __bf16* __restrict__ wi,     // [L,H,H] bf16
                      const float*  __restrict__ bfv,    // [L,H]
                      const float*  __restrict__ biv,    // [L,H]
                      float*        __restrict__ out,    // [T,B,H] spikes
                      float*        __restrict__ states, // [L,B,H] h (reset)
                      __bf16*       __restrict__ ybuf,   // 2*[B,H] ping-pong
                      unsigned*     __restrict__ bar) {
  // (L*2 gates) * 32 rows * RSTRIDE bf16 = 266,240 bytes of LDS
  __shared__ __bf16 smem[L_DIM * 2 * 32 * RSTRIDE];

  const int lane  = threadIdx.x & 31;
  const int wv    = threadIdx.x >> 5;     // wave in block: 0/1
  const int m     = blockIdx.x >> 4;      // 0..3   M tile (batch rows)
  const int npair = blockIdx.x & 15;      // 0..15  pair of N tiles
  const int n     = npair * 2 + wv;       // 0..31  this wave's N tile
  const int lrow  = lane & 15;
  const int khalf = (lane >> 4) << 3;                // 0 or 8
  const int rbase = m * 16 + ((lane >> 4) << 3);     // C/D layout M base
  const int gn    = n * 16 + lrow;                   // this lane's N column
  const int arow  = m * 16 + lrow;                   // this lane's A row

  // ---- stage weight slice into LDS (once; amortized over 512 steps) ----
  // 256 logical rows: [layer(4)][gate(2)][row(32)] of 512 bf16 each.
  for (int r = threadIdx.x; r < L_DIM * 2 * 32; r += blockDim.x) {
    const int l  = r >> 6;
    const int g  = (r >> 5) & 1;
    const int rr = r & 31;
    const __bf16* src = (g ? wi : wf) + (size_t)l * H_DIM * H_DIM +
                        (size_t)(npair * 32 + rr) * H_DIM;
    __bf16* dst = smem + (size_t)r * RSTRIDE;
#pragma unroll 4
    for (int j = 0; j < H_DIM; j += 8) {           // 16B vector copies
      *(uint4*)(dst + j) = *(const uint4*)(src + j);
    }
  }
  __syncthreads();

  unsigned round = 0;

  for (int t = 0; t < T_DIM; ++t) {
    const float* xt = x + (size_t)t * (B_DIM * H_DIM);

    for (int l = 0; l < L_DIM; ++l) {
      // layer l reads ping buffer ((l&1)^1), writes pong buffer (l&1);
      // layer 0 reads x[t] directly (f32 -> bf16 on the fly).
      const __bf16* yin  = ybuf + ((l & 1) ^ 1) * (B_DIM * H_DIM);
      __bf16*       yout = ybuf + (l & 1) * (B_DIM * H_DIM);

      // LDS fragment row bases for this lane (N column = lrow within tile)
      const __bf16* bF = smem + (size_t)((l * 2 + 0) * 32 + wv * 16 + lrow) * RSTRIDE;
      const __bf16* bI = smem + (size_t)((l * 2 + 1) * 32 + wv * 16 + lrow) * RSTRIDE;

      v8f accF = {};
      v8f accI = {};

      if (l == 0) {
        const float* xr = xt + (size_t)arow * H_DIM;
#pragma unroll 4
        for (int kb = 0; kb < 16; ++kb) {
          const int k0 = kb * 32 + khalf;
          __builtin_prefetch(xr + k0 + 128, 0, 3);
          v16bf a  = load_frag_f32(xr + k0);
          v16bf b0 = load_frag_bf16(bF + k0);      // ds_load_b128 x2
          v16bf b1 = load_frag_bf16(bI + k0);
          accF = __builtin_amdgcn_wmma_f32_16x16x32_bf16(
              false, a, false, b0, (short)0, accF, false, false);
          accI = __builtin_amdgcn_wmma_f32_16x16x32_bf16(
              false, a, false, b1, (short)0, accI, false, false);
        }
      } else {
        const __bf16* yr = yin + (size_t)arow * H_DIM;
#pragma unroll 4
        for (int kb = 0; kb < 16; ++kb) {
          const int k0 = kb * 32 + khalf;
          v16bf a  = load_frag_bf16(yr + k0);      // global_load_b128 x2
          v16bf b0 = load_frag_bf16(bF + k0);      // ds_load_b128 x2
          v16bf b1 = load_frag_bf16(bI + k0);
          accF = __builtin_amdgcn_wmma_f32_16x16x32_bf16(
              false, a, false, b0, (short)0, accF, false, false);
          accI = __builtin_amdgcn_wmma_f32_16x16x32_bf16(
              false, a, false, b1, (short)0, accI, false, false);
        }
      }

      // ---- fused LIF epilogue on the 16x16 f32 C tile ----
      const float bFc = bfv[l * H_DIM + gn];
      const float bIc = biv[l * H_DIM + gn];
      float* st = states + (size_t)l * (B_DIM * H_DIM);

#pragma unroll
      for (int i = 0; i < 8; ++i) {
        const int    gm  = rbase + i;                 // batch row of this elem
        const size_t off = (size_t)gm * H_DIM + gn;
        const float F  = 1.0f / (1.0f + __expf(-(accF[i] + bFc)));  // sigmoid
        float       Iv = accI[i] + bIc;                             // relu
        Iv = Iv > 0.0f ? Iv : 0.0f;
        const float h  = F * st[off] + (1.0f - F) * Iv;             // leak/input
        const float s  = (h >= 0.0f) ? 1.0f : 0.0f;                 // spike
        const float hr = (1.0f - s) * h;                            // reset
        st[off]   = hr;                        // persistent state / final_states
        yout[off] = (__bf16)hr;                // next layer's input
        if (l == L_DIM - 1) {
          out[(size_t)t * (B_DIM * H_DIM) + off] = s;
        }
      }

      ++round;
      grid_barrier(bar, round);   // publish y / state before next layer/step
    }
  }
}

// ---------------------------------------------------------------------------
// Host launcher. d_in order: x, Wf, bf, Wi, bi.
// d_out = out[T,B,H] ++ final_states[L,B,H] (f32).
// d_ws layout: [0,256) barrier; then Wf bf16 (2MB); Wi bf16 (2MB); y ping-pong.
// ---------------------------------------------------------------------------
extern "C" void kernel_launch(void* const* d_in, const int* in_sizes, int n_in,
                              void* d_out, int out_size, void* d_ws, size_t ws_size,
                              hipStream_t stream) {
  const float* x  = (const float*)d_in[0];
  const float* Wf = (const float*)d_in[1];
  const float* bf = (const float*)d_in[2];
  const float* Wi = (const float*)d_in[3];
  const float* bi = (const float*)d_in[4];

  float* out    = (float*)d_out;
  float* states = out + (size_t)T_DIM * B_DIM * H_DIM;   // state lives in d_out tail

  uintptr_t ws   = (uintptr_t)d_ws;
  unsigned* bar  = (unsigned*)ws;
  __bf16*   wf16 = (__bf16*)(ws + 256);
  __bf16*   wi16 = wf16 + (size_t)L_DIM * H_DIM * H_DIM;
  __bf16*   ybuf = wi16 + (size_t)L_DIM * H_DIM * H_DIM;

  lif_init_kernel<<<1024, 256, 0, stream>>>(Wf, Wi, wf16, wi16, states, bar);
  lif_persistent_kernel<<<NBLOCKS, 64, 0, stream>>>(
      x, wf16, wi16, bf, bi, out, states, ybuf, bar);
}
